// Transfusion_17995912970859
// MI455X (gfx1250) — matrix operand order repaired and checked by reading, not compiled, not run on hardware
//
#include <hip/hip_runtime.h>
#include <hip/hip_bf16.h>
#include <math.h>

// ---------------------------------------------------------------------------
// Transfusion attention layer for MI455X (gfx1250), wave32 + WMMA bf16
// + async global->LDS staging (ASYNCcnt) with LDS double buffering.
// ---------------------------------------------------------------------------

typedef __attribute__((ext_vector_type(16))) __bf16 v16bf;
typedef __attribute__((ext_vector_type(8)))  float  v8f;

#define BATCH   2
#define SEQ     2048
#define DIM     2048
#define HEADS   16
#define DHEAD   64
#define DINNER  1024     // HEADS * DHEAD
#define NQKV    3072     // 3 * DINNER
#define SOFTCAPF 50.0f

union AFrag {
    v16bf v;
    unsigned short u[16];
    unsigned int   d[8];
};

__device__ __forceinline__ unsigned short f2bf(float f) {
    unsigned int u = __float_as_uint(f);
    u = (u + 0x7FFFu + ((u >> 16) & 1u)) >> 16;   // round-to-nearest-even
    return (unsigned short)u;
}

// Async 16B global -> LDS copy, no VGPR round-trip (tracked by ASYNCcnt).
// LDS operand is the wave-relative AS(3) byte address == low 32 bits of the
// generic pointer to a __shared__ object.
__device__ __forceinline__ void async_g2l_b128(void* lds_ptr, const void* gptr) {
    unsigned lds = (unsigned)(size_t)lds_ptr;
    asm volatile("global_load_async_to_lds_b128 %0, %1, off"
                 :: "v"(lds), "v"(gptr) : "memory");
}
__device__ __forceinline__ void wait_async0() {
    asm volatile("s_wait_asynccnt 0" ::: "memory");
}

// ---------------------------------------------------------------------------
// Kernel 1: RMSNorm (f32 -> bf16).  One block per row of [B*S, DIM].
// ---------------------------------------------------------------------------
__global__ void k_rmsnorm_bf16(const float* __restrict__ x,
                               const float* __restrict__ gamma,
                               unsigned short* __restrict__ xnb) {
    const int row = blockIdx.x;          // 0..B*S-1
    const int tid = threadIdx.x;         // 256 threads
    const float* xr = x + (size_t)row * DIM;

    float vals[8];
    float ss = 0.f;
#pragma unroll
    for (int i = 0; i < 8; ++i) {
        float v = xr[tid + i * 256];
        vals[i] = v;
        ss += v * v;
    }
#pragma unroll
    for (int m = 1; m < 32; m <<= 1) ss += __shfl_xor(ss, m, 32);

    __shared__ float red[8];
    if ((tid & 31) == 0) red[tid >> 5] = ss;
    __syncthreads();
    float tot = 0.f;
#pragma unroll
    for (int i = 0; i < 8; ++i) tot += red[i];

    const float n   = sqrtf(tot);
    const float inv = 45.254833995939045f / fmaxf(n, 1e-12f);  // sqrt(2048)/max(n,eps)

#pragma unroll
    for (int i = 0; i < 8; ++i) {
        int c = tid + i * 256;
        xnb[(size_t)row * DIM + c] = f2bf(vals[i] * inv * (gamma[c] + 1.0f));
    }
}

// ---------------------------------------------------------------------------
// Kernel 2: f32 -> bf16 elementwise (weights).
// ---------------------------------------------------------------------------
__global__ void k_f32_to_bf16(const float* __restrict__ src,
                              unsigned short* __restrict__ dst, int n) {
    int i = blockIdx.x * 256 + threadIdx.x;
    const int stride = gridDim.x * 256;
    for (; i < n; i += stride) dst[i] = f2bf(src[i]);
}

// ---------------------------------------------------------------------------
// WMMA GEMM:  C[M,N] = A[M,K](bf16) * B[K,N](bf16), f32 accumulate.
// 256 threads = 8 waves; block tile 128x128, K-step 32, double-buffered LDS.
// A tile staged with global_load_async_to_lds_b128 (no VGPR round-trip);
// B tile transposed manually (async copy cannot transpose).
// Wave grid 4(M) x 2(N); each wave owns 32x64 = 2x4 WMMA tiles.
// MODE 0: QKV epilogue (scale Q, scatter to q/k/v [B,H,S,64] bf16)
// MODE 1: plain f32 store.
// ---------------------------------------------------------------------------
template <int M, int N, int K, int MODE>
__global__ void k_gemm_bf16(const unsigned short* __restrict__ A,
                            const unsigned short* __restrict__ B,
                            unsigned short* __restrict__ Oq,
                            unsigned short* __restrict__ Ok,
                            unsigned short* __restrict__ Ov,
                            float* __restrict__ Of) {
    const int tid   = threadIdx.x;
    const int lane  = tid & 31;
    const int wave  = tid >> 5;           // 0..7
    const int wm    = wave & 3;           // 4 waves along M
    const int wn    = wave >> 2;          // 2 waves along N
    const int m0    = blockIdx.y * 128;
    const int n0    = blockIdx.x * 128;
    const int lrow  = lane & 15;
    const int lhalf = lane >> 4;

    __shared__ unsigned short As[2][128][48];   // [buf][m][k], 96B pitch
    __shared__ unsigned short Bs[2][128][48];   // [buf][n][k] (transposed fill)

    v8f acc[2][4];
#pragma unroll
    for (int mi = 0; mi < 2; ++mi)
#pragma unroll
        for (int ni = 0; ni < 4; ++ni)
            acc[mi][ni] = (v8f){0.f, 0.f, 0.f, 0.f, 0.f, 0.f, 0.f, 0.f};

    // ---- tile stagers -----------------------------------------------------
    auto stageA = [&](int kk, int buf) {
#pragma unroll
        for (int t = 0; t < 2; ++t) {
            int idx = tid + t * 256;                // 0..511
            int r = idx >> 2, seg = idx & 3;
            async_g2l_b128(&As[buf][r][seg * 8],
                           A + (size_t)(m0 + r) * K + kk + seg * 8);
        }
    };
    auto stageB = [&](int kk, int buf) {
#pragma unroll
        for (int t = 0; t < 2; ++t) {
            int idx = tid + t * 256;                // 0..511
            int kr = idx >> 4, nseg = idx & 15;
            uint4 v = *(const uint4*)(B + (size_t)(kk + kr) * N + n0 + nseg * 8);
            const unsigned short* pu = (const unsigned short*)&v;
#pragma unroll
            for (int e = 0; e < 8; ++e) Bs[buf][nseg * 8 + e][kr] = pu[e];
        }
    };

    stageA(0, 0);
    stageB(0, 0);
    wait_async0();
    __syncthreads();

    int cur = 0;
    for (int kk = 0; kk < K; kk += 32) {
        // prefetch next K-slab into the other buffer while we compute
        if (kk + 32 < K) {
            stageA(kk + 32, cur ^ 1);
            stageB(kk + 32, cur ^ 1);
        }

        // ---- fragments from LDS (current buffer)
        AFrag a[2];
#pragma unroll
        for (int mi = 0; mi < 2; ++mi) {
            int r  = wm * 32 + mi * 16 + lrow;
            int kb = lhalf * 8;
            const unsigned int* p = (const unsigned int*)&As[cur][r][0];
#pragma unroll
            for (int e = 0; e < 4; ++e) a[mi].d[e]     = p[(kb >> 1) + e];
#pragma unroll
            for (int e = 0; e < 4; ++e) a[mi].d[4 + e] = p[((kb + 16) >> 1) + e];
        }
        AFrag bfr[4];
#pragma unroll
        for (int ni = 0; ni < 4; ++ni) {
            int c  = wn * 64 + ni * 16 + lrow;
            int kb = lhalf * 8;
            const unsigned int* p = (const unsigned int*)&Bs[cur][c][kb];
#pragma unroll
            for (int e = 0; e < 8; ++e) bfr[ni].d[e] = p[e < 4 ? e : e + 4];
        }
#pragma unroll
        for (int mi = 0; mi < 2; ++mi)
#pragma unroll
            for (int ni = 0; ni < 4; ++ni)
                acc[mi][ni] = __builtin_amdgcn_wmma_f32_16x16x32_bf16(
                    false, a[mi].v, false, bfr[ni].v,
                    (short)0, acc[mi][ni], false, false);

        wait_async0();        // our async writes into buf^1 are done
        __syncthreads();      // everyone done reading cur / writing cur^1
        cur ^= 1;
    }

    // ---- epilogue (C layout: elem e -> row e + 8*lhalf, col = lrow)
#pragma unroll
    for (int mi = 0; mi < 2; ++mi) {
#pragma unroll
        for (int ni = 0; ni < 4; ++ni) {
#pragma unroll
            for (int e = 0; e < 8; ++e) {
                int m = m0 + wm * 32 + mi * 16 + e + lhalf * 8;
                int n = n0 + wn * 64 + ni * 16 + lrow;
                float v = acc[mi][ni][e];
                if (MODE == 0) {
                    int which = n >> 10;            // 0=q 1=k 2=v
                    int nn = n & 1023;
                    int h = nn >> 6, dd = nn & 63;
                    int b = m >> 11, s = m & 2047;
                    size_t off = (((size_t)(b * HEADS + h) * SEQ) + s) * DHEAD + dd;
                    unsigned short w16 = f2bf(which == 0 ? v * 0.125f : v);  // Q * d^-1/2
                    (which == 0 ? Oq : (which == 1 ? Ok : Ov))[off] = w16;
                } else {
                    Of[(size_t)m * N + n] = v;
                }
            }
        }
    }
}

// ---------------------------------------------------------------------------
// Kernel 4: flash attention with tanh softcap + transfusion mask.
// Block = (b, h, 128 queries); 8 waves, 16 query rows per wave.
// KV tiles of 64; K staged [key][d] via async-to-LDS, V transposed manually.
// ---------------------------------------------------------------------------
__global__ void k_attn(const unsigned short* __restrict__ Q,
                       const unsigned short* __restrict__ Kg,
                       const unsigned short* __restrict__ Vg,
                       const int* __restrict__ mods,
                       unsigned short* __restrict__ O) {
    const int tid = threadIdx.x, lane = tid & 31, wave = tid >> 5;
    const int lrow = lane & 15, lhalf = lane >> 4;
    const int nqt = SEQ / 128;                 // 16 query tiles
    const int qt  = blockIdx.x % nqt;
    const int bh  = blockIdx.x / nqt;
    const int b   = bh / HEADS, h = bh % HEADS;

    const int off0 = mods[b * 6 + 1], end0 = off0 + mods[b * 6 + 2];
    const int off1 = mods[b * 6 + 4], end1 = off1 + mods[b * 6 + 5];

    const size_t base = (size_t)bh * SEQ * DHEAD;
    const int s0   = qt * 128 + wave * 16;
    const int q_hi = qt * 128 + 127;

    __shared__ unsigned short Ks[64][72];       // [key][feature]
    __shared__ unsigned short Vt[64][72];       // [feature d][key]
    __shared__ unsigned short Ps[8][16][72];    // per-wave P scratch [row][key]

    // Q A-fragments (2 chunks of K=32 features), straight from global
    AFrag qf[2];
#pragma unroll
    for (int f = 0; f < 2; ++f) {
        const unsigned short* qp = Q + base + (size_t)(s0 + lrow) * DHEAD + f * 32 + lhalf * 8;
        ((uint4*)qf[f].d)[0] = *(const uint4*)qp;
        ((uint4*)qf[f].d)[1] = *(const uint4*)(qp + 16);
    }

    v8f acc[4];
#pragma unroll
    for (int ni = 0; ni < 4; ++ni) acc[ni] = (v8f){0.f,0.f,0.f,0.f,0.f,0.f,0.f,0.f};
    float mrow[8], lsum[8];
#pragma unroll
    for (int e = 0; e < 8; ++e) { mrow[e] = -3.0e38f; lsum[e] = 0.f; }

    for (int j0 = 0; j0 < SEQ; j0 += 64) {
        // uniform tile-skip: causal or any modality span can reach this tile
        bool needed = (j0 <= q_hi) ||
                      (q_hi >= off0 && j0 < end0) ||
                      (q_hi >= off1 && j0 < end1);
        if (!needed) continue;

        // ---- K tile: straight async copy global -> LDS
#pragma unroll
        for (int t = 0; t < 2; ++t) {
            int idx = tid + t * 256;            // 0..511
            int key = idx >> 3, seg = idx & 7;
            async_g2l_b128(&Ks[key][seg * 8],
                           Kg + base + (size_t)(j0 + key) * DHEAD + seg * 8);
        }
        // ---- V tile: manual transpose into [d][key]
#pragma unroll
        for (int t = 0; t < 2; ++t) {
            int idx = tid + t * 256;
            int key = idx >> 3, seg = idx & 7;
            uint4 vv = *(const uint4*)(Vg + base + (size_t)(j0 + key) * DHEAD + seg * 8);
            const unsigned short* pu = (const unsigned short*)&vv;
#pragma unroll
            for (int e = 0; e < 8; ++e) Vt[seg * 8 + e][key] = pu[e];
        }
        wait_async0();
        __syncthreads();

        // ---- S = Q K^T  (4 subtiles of 16 keys, 2 WMMAs each over d=64)
        v8f s[4];
#pragma unroll
        for (int js = 0; js < 4; ++js) {
            int key = js * 16 + lrow;
            AFrag kb0, kb1;
            {
                const unsigned int* p0 = (const unsigned int*)&Ks[key][lhalf * 8];
                const unsigned int* p1 = (const unsigned int*)&Ks[key][32 + lhalf * 8];
#pragma unroll
                for (int e = 0; e < 8; ++e) { kb0.d[e] = p0[e < 4 ? e : e + 4]; kb1.d[e] = p1[e < 4 ? e : e + 4]; }
            }
            v8f z = (v8f){0.f,0.f,0.f,0.f,0.f,0.f,0.f,0.f};
            z = __builtin_amdgcn_wmma_f32_16x16x32_bf16(false, qf[0].v, false, kb0.v,
                                                        (short)0, z, false, false);
            z = __builtin_amdgcn_wmma_f32_16x16x32_bf16(false, qf[1].v, false, kb1.v,
                                                        (short)0, z, false, false);
            s[js] = z;
        }

        // ---- softcap + mask
#pragma unroll
        for (int js = 0; js < 4; ++js) {
            int j = j0 + js * 16 + lrow;
#pragma unroll
            for (int e = 0; e < 8; ++e) {
                int i = s0 + e + lhalf * 8;
                float sv = tanhf(s[js][e] * (1.0f / SOFTCAPF)) * SOFTCAPF;
                bool ok = (i >= j) || (i >= off0 && j < end0) || (i >= off1 && j < end1);
                s[js][e] = ok ? sv : -3.0e38f;
            }
        }

        // ---- online softmax: tile row max across 16 lanes of each half-wave
        float scale[8];
#pragma unroll
        for (int e = 0; e < 8; ++e) {
            float v = fmaxf(fmaxf(s[0][e], s[1][e]), fmaxf(s[2][e], s[3][e]));
            v = fmaxf(v, __shfl_xor(v, 1, 32));
            v = fmaxf(v, __shfl_xor(v, 2, 32));
            v = fmaxf(v, __shfl_xor(v, 4, 32));
            v = fmaxf(v, __shfl_xor(v, 8, 32));
            float mn = fmaxf(mrow[e], v);
            scale[e] = __expf(mrow[e] - mn);
            mrow[e]  = mn;
        }

        // ---- P = exp(s - m); stash bf16 P in LDS for A-fragment re-layout
        float ps[8];
#pragma unroll
        for (int e = 0; e < 8; ++e) ps[e] = 0.f;
#pragma unroll
        for (int js = 0; js < 4; ++js) {
#pragma unroll
            for (int e = 0; e < 8; ++e) {
                float p = __expf(s[js][e] - mrow[e]);
                ps[e] += p;
                Ps[wave][e + lhalf * 8][js * 16 + lrow] = f2bf(p);
            }
        }
#pragma unroll
        for (int e = 0; e < 8; ++e) {
            float v = ps[e];
            v += __shfl_xor(v, 1, 32);
            v += __shfl_xor(v, 2, 32);
            v += __shfl_xor(v, 4, 32);
            v += __shfl_xor(v, 8, 32);
            lsum[e] = lsum[e] * scale[e] + v;
        }
#pragma unroll
        for (int ni = 0; ni < 4; ++ni)
#pragma unroll
            for (int e = 0; e < 8; ++e)
                acc[ni][e] *= scale[e];

        asm volatile("s_wait_dscnt 0" ::: "memory");   // wave-local LDS RAW

        // ---- acc += P @ V
        AFrag pf[2];
#pragma unroll
        for (int f = 0; f < 2; ++f) {
            const unsigned int* p = (const unsigned int*)&Ps[wave][lrow][0];
            int kb = f * 32 + lhalf * 8;
#pragma unroll
            for (int e = 0; e < 4; ++e) pf[f].d[e]     = p[(kb >> 1) + e];
#pragma unroll
            for (int e = 0; e < 4; ++e) pf[f].d[4 + e] = p[((kb + 16) >> 1) + e];
        }
#pragma unroll
        for (int ni = 0; ni < 4; ++ni) {
            int dd = ni * 16 + lrow;
            AFrag vb0, vb1;
            {
                const unsigned int* p0 = (const unsigned int*)&Vt[dd][lhalf * 8];
                const unsigned int* p1 = (const unsigned int*)&Vt[dd][32 + lhalf * 8];
#pragma unroll
                for (int e = 0; e < 8; ++e) { vb0.d[e] = p0[e < 4 ? e : e + 4]; vb1.d[e] = p1[e < 4 ? e : e + 4]; }
            }
            acc[ni] = __builtin_amdgcn_wmma_f32_16x16x32_bf16(false, pf[0].v, false, vb0.v,
                                                              (short)0, acc[ni], false, false);
            acc[ni] = __builtin_amdgcn_wmma_f32_16x16x32_bf16(false, pf[1].v, false, vb1.v,
                                                              (short)0, acc[ni], false, false);
        }
        __syncthreads();
    }

    // ---- normalize and store bf16 into [B, S, H*DHEAD]
#pragma unroll
    for (int e = 0; e < 8; ++e) {
        float rinv = 1.0f / fmaxf(lsum[e], 1e-20f);
        int srow = s0 + e + lhalf * 8;
        size_t orow = ((size_t)b * SEQ + srow) * DINNER + (size_t)h * DHEAD;
#pragma unroll
        for (int ni = 0; ni < 4; ++ni)
            O[orow + ni * 16 + lrow] = f2bf(acc[ni][e] * rinv);
    }
}

// ---------------------------------------------------------------------------
// Host-side launcher.
// ---------------------------------------------------------------------------
extern "C" void kernel_launch(void* const* d_in, const int* in_sizes, int n_in,
                              void* d_out, int out_size, void* d_ws, size_t ws_size,
                              hipStream_t stream) {
    (void)in_sizes; (void)n_in; (void)out_size; (void)ws_size;
    const float* x     = (const float*)d_in[0];    // [2,2048,2048]
    const float* gamma = (const float*)d_in[1];    // [2048]
    const float* wqkv  = (const float*)d_in[2];    // [2048,3072]
    const float* wout  = (const float*)d_in[3];    // [1024,2048]
    const int*   mods  = (const int*)d_in[4];      // [2,2,3]
    float* out = (float*)d_out;

    char* ws = (char*)d_ws;
    unsigned short* xnb   = (unsigned short*)(ws);                 // 16 MiB
    unsigned short* wqkvb = (unsigned short*)(ws + (16u << 20));   // 12 MiB
    unsigned short* woutb = (unsigned short*)(ws + (28u << 20));   //  4 MiB
    unsigned short* qb    = (unsigned short*)(ws + (32u << 20));   //  8 MiB
    unsigned short* kb    = (unsigned short*)(ws + (40u << 20));   //  8 MiB
    unsigned short* vb    = (unsigned short*)(ws + (48u << 20));   //  8 MiB
    unsigned short* aob   = (unsigned short*)(ws + (56u << 20));   //  8 MiB

    // 1) RMSNorm -> bf16
    k_rmsnorm_bf16<<<BATCH * SEQ, 256, 0, stream>>>(x, gamma, xnb);

    // 2) weights -> bf16
    k_f32_to_bf16<<<2048, 256, 0, stream>>>(wqkv, wqkvb, DIM * NQKV);
    k_f32_to_bf16<<<1024, 256, 0, stream>>>(wout, woutb, DINNER * DIM);

    // 3) QKV projection (M=4096, N=3072, K=2048), scatter to q/k/v
    k_gemm_bf16<BATCH * SEQ, NQKV, DIM, 0>
        <<<dim3(NQKV / 128, (BATCH * SEQ) / 128), 256, 0, stream>>>(
            xnb, wqkvb, qb, kb, vb, nullptr);

    // 4) flash attention: B*H*(S/128) blocks
    k_attn<<<BATCH * HEADS * (SEQ / 128), 256, 0, stream>>>(qb, kb, vb, mods, aob);

    // 5) output projection (M=4096, N=2048, K=1024) -> f32 d_out
    k_gemm_bf16<BATCH * SEQ, DIM, DINNER, 1>
        <<<dim3(DIM / 128, (BATCH * SEQ) / 128), 256, 0, stream>>>(
            aob, woutb, nullptr, nullptr, nullptr, out);
}
